// CaptioningRNN_21277267984517
// MI455X (gfx1250) — matrix operand order repaired and checked by reading, not compiled, not run on hardware
//
#include <hip/hip_runtime.h>
#include <hip/hip_bf16.h>

// ---------------------------------------------------------------------------
// CDNA5 (gfx1250) CaptioningRNN forward loss, bf16 WMMA everywhere.
// Dims (fixed by reference): N=256, T=64, D=1280, W=256, H=512, V=10000
// ---------------------------------------------------------------------------

typedef __attribute__((ext_vector_type(16))) __bf16 v16bf;
typedef __attribute__((ext_vector_type(8)))  float  v8f;

#define CN 256
#define CT 64
#define CD 1280
#define CW 256
#define CH 512
#define CV 10000

__device__ __forceinline__ v8f wmma_bf16f32(v16bf a, v16bf b, v8f c) {
  // D = A(16x32 bf16) * B(32x16 bf16) + C(16x16 f32)
  return __builtin_amdgcn_wmma_f32_16x16x32_bf16(
      /*neg_a=*/false, a, /*neg_b=*/false, b,
      /*c_mod=*/(short)0, c, /*reuse_a=*/false, /*reuse_b=*/false);
}

// Zero the output scalar(s).
__global__ void zero_f32_kernel(float* p, int n) {
  int i = blockIdx.x * blockDim.x + threadIdx.x;
  if (i < n) p[i] = 0.0f;
}

// out[N x K] (bf16) = transpose(in[K x N] f32)
__global__ void transpose_to_bf16_kernel(const float* __restrict__ in,
                                         __bf16* __restrict__ out,
                                         int K, int N) {
  long long i = (long long)blockIdx.x * blockDim.x + threadIdx.x;
  long long total = (long long)K * N;
  if (i < total) {
    int n = (int)(i / K);
    int k = (int)(i % K);
    out[i] = (__bf16)in[(size_t)k * N + n];
  }
}

// ---------------------------------------------------------------------------
// Generic wave-per-16x16-tile GEMM:  C[M x N] = A[M x K](f32) * Bt[N x K]^T + bias
// If caps != nullptr, row m of A is gathered: A_row = A + caps[(m/T)*(T+1)+m%T]*K
// (embedding lookup fused into the GEMM).
// ---------------------------------------------------------------------------
__global__ void gemm_af32_btbf16_kernel(const float* __restrict__ A,
                                        const int*   __restrict__ caps,
                                        const __bf16* __restrict__ Bt,
                                        const float* __restrict__ bias,
                                        float* __restrict__ C,
                                        int M, int N, int K) {
  const int wavesPerBlock = blockDim.x >> 5;
  int waveId = blockIdx.x * wavesPerBlock + (threadIdx.x >> 5);
  int tilesN = N >> 4;
  int totalTiles = (M >> 4) * tilesN;
  if (waveId >= totalTiles) return;         // whole-wave exit: EXEC stays all-1s
  int tm = waveId / tilesN;
  int tn = waveId % tilesN;
  int lane = threadIdx.x & 31;
  int g  = lane >> 4;                        // lane group (ISA A/B/C layouts)
  int ln = lane & 15;

  int mrow = tm * 16 + ln;                   // A row owned by this lane
  const float* arow;
  if (caps) {
    int n = mrow / CT, t = mrow % CT;        // flat row = n*T + t
    arow = A + (size_t)caps[n * (CT + 1) + t] * K;
  } else {
    arow = A + (size_t)mrow * K;
  }
  int ncol = tn * 16 + ln;                   // B column owned by this lane
  const __bf16* brow = Bt + (size_t)ncol * K;

  v8f acc;
  float bv = bias ? bias[ncol] : 0.0f;
#pragma unroll
  for (int r = 0; r < 8; ++r) acc[r] = bv;

  for (int k0 = 0; k0 < K; k0 += 32) {
    v16bf a, b;
    const float* a0 = arow + k0 + 8 * g;        // K = k0 + 8g .. +7
    const float* a1 = arow + k0 + 16 + 8 * g;   // K = k0 + 16 + 8g .. +7
#pragma unroll
    for (int i = 0; i < 8; ++i) { a[i] = (__bf16)a0[i]; a[8 + i] = (__bf16)a1[i]; }
    const __bf16* bp = brow + k0 + 8 * g;       // K = k0 + 8g .. +7, k0 + 16 + 8g .. +7
#pragma unroll
    for (int i = 0; i < 8; ++i) { b[i] = bp[i]; b[8 + i] = bp[16 + i]; }
    acc = wmma_bf16f32(a, b, acc);
  }
#pragma unroll
  for (int r = 0; r < 8; ++r) {
    int row = tm * 16 + r + 8 * g;              // D: element r -> row r+8g, col ln
    C[(size_t)row * N + ncol] = acc[r];
  }
}

// ---------------------------------------------------------------------------
// One RNN step: h_next = tanh(h_prev @ Wh + xWxb[:, t, :])
// M = 256 (batch), N = K = 512. Writes f32 h_next (ping-pong) and bf16 h for
// the fused scores stage.
// ---------------------------------------------------------------------------
__global__ void rnn_step_kernel(const float* __restrict__ hprev,
                                const __bf16* __restrict__ Wh_t,
                                const float* __restrict__ xWxb,
                                float* __restrict__ hnext,
                                __bf16* __restrict__ hs_bf,
                                int t) {
  int waveId = blockIdx.x * (blockDim.x >> 5) + (threadIdx.x >> 5);
  if (waveId >= (CN / 16) * (CH / 16)) return;
  int tm = waveId / (CH / 16);
  int tn = waveId % (CH / 16);
  int lane = threadIdx.x & 31;
  int g = lane >> 4, ln = lane & 15;
  int mrow = tm * 16 + ln;
  int ncol = tn * 16 + ln;
  const float*  arow = hprev + (size_t)mrow * CH;
  const __bf16* brow = Wh_t + (size_t)ncol * CH;

  v8f acc;
#pragma unroll
  for (int r = 0; r < 8; ++r) {
    int row = tm * 16 + r + 8 * g;
    acc[r] = xWxb[((size_t)row * CT + t) * CH + ncol];   // pre-add x@Wx + b
  }
  for (int k0 = 0; k0 < CH; k0 += 32) {
    v16bf a, b;
    const float* a0 = arow + k0 + 8 * g;
    const float* a1 = arow + k0 + 16 + 8 * g;
#pragma unroll
    for (int i = 0; i < 8; ++i) { a[i] = (__bf16)a0[i]; a[8 + i] = (__bf16)a1[i]; }
    const __bf16* bp = brow + k0 + 8 * g;
#pragma unroll
    for (int i = 0; i < 8; ++i) { b[i] = bp[i]; b[8 + i] = bp[16 + i]; }
    acc = wmma_bf16f32(a, b, acc);
  }
#pragma unroll
  for (int r = 0; r < 8; ++r) {
    int row = tm * 16 + r + 8 * g;
    float h = tanhf(acc[r]);
    hnext[(size_t)row * CH + ncol] = h;
    hs_bf[((size_t)row * CT + t) * CH + ncol] = (__bf16)h;
  }
}

// ---------------------------------------------------------------------------
// Fused scores + log_softmax + masked NLL.
// Block = 256 threads (8 waves), one block per 16-row strip of h (1024 blocks).
// A strip (16 x 512 bf16 = 16 KB) staged in LDS; each wave processes PAIRS of
// 16-wide V tiles (313 pairs over V=10000) so one A fragment feeds two WMMAs,
// halving LDS traffic per WMMA. Online logsumexp merges both tiles' columns
// before the cross-lane reduce. Target score computed from the same bf16 data
// for consistency. One atomicAdd per block.
// ---------------------------------------------------------------------------
__global__ void scores_loss_kernel(const __bf16* __restrict__ hs,
                                   const __bf16* __restrict__ Wout_t,
                                   const float* __restrict__ b_out,
                                   const int*   __restrict__ captions,
                                   float* __restrict__ loss) {
  __shared__ __align__(16) __bf16 As[16 * CH];
  __shared__ float red[8][16][2];
  __shared__ float lse[16];
  __shared__ float part[16][16];
  __shared__ float rowNll[16];

  int tid   = threadIdx.x;
  int strip = blockIdx.x;                          // rows strip*16 .. +15

  const __bf16* src = hs + (size_t)strip * 16 * CH;
  for (int j = tid; j < 16 * CH; j += 256) As[j] = src[j];
  __syncthreads();

  int wave = tid >> 5, lane = tid & 31;
  int g = lane >> 4, ln = lane & 15;

  float rmax[8], rsum[8];
#pragma unroll
  for (int r = 0; r < 8; ++r) { rmax[r] = -1e30f; rsum[r] = 0.0f; }

  const __bf16* a_base = As + (size_t)ln * CH;
  const int NPAIRS = (CV / 16 + 1) / 2;            // 313 pairs of 16-col tiles
  for (int tp = wave; tp < NPAIRS; tp += 8) {      // wave-uniform loop
    int n0 = tp * 32;
    bool has2 = (n0 + 16) < CV;                    // last pair covers 1 tile

    // Prefetch this wave's next pair of W_out rows (256 rows = 256KB ahead).
    if (n0 + 256 < CV) {
      __builtin_prefetch(Wout_t + (size_t)(n0 + 256 + ln) * CH, 0, 1);
      __builtin_prefetch(Wout_t + (size_t)(n0 + 256 + 16 + ln) * CH, 0, 1);
    }

    v8f acc0, acc1;
    float bv0 = b_out[n0 + ln];
    float bv1 = has2 ? b_out[n0 + 16 + ln] : 0.0f;
#pragma unroll
    for (int r = 0; r < 8; ++r) { acc0[r] = bv0; acc1[r] = bv1; }
    const __bf16* brow0 = Wout_t + (size_t)(n0 + ln) * CH;
    const __bf16* brow1 = Wout_t + (size_t)(n0 + 16 + ln) * CH;

    for (int k0 = 0; k0 < CH; k0 += 32) {
      v16bf a, b0;
      const __bf16* a0 = a_base + k0 + 8 * g;
      const __bf16* a1 = a_base + k0 + 16 + 8 * g;
#pragma unroll
      for (int i = 0; i < 8; ++i) { a[i] = a0[i]; a[8 + i] = a1[i]; }
      const __bf16* bp0 = brow0 + k0 + 8 * g;
#pragma unroll
      for (int i = 0; i < 8; ++i) { b0[i] = bp0[i]; b0[8 + i] = bp0[16 + i]; }
      acc0 = wmma_bf16f32(a, b0, acc0);
      if (has2) {                                  // wave-uniform branch
        v16bf b1;
        const __bf16* bp1 = brow1 + k0 + 8 * g;
#pragma unroll
        for (int i = 0; i < 8; ++i) { b1[i] = bp1[i]; b1[8 + i] = bp1[16 + i]; }
        acc1 = wmma_bf16f32(a, b1, acc1);          // A fragment reused
      }
    }

    // Row m = r + 8g lives on acc[r] across the 16 lanes of group g.
    // Merge both tiles' columns per lane first, then one cross-lane reduce.
#pragma unroll
    for (int r = 0; r < 8; ++r) {
      float v0 = acc0[r];
      float v1 = has2 ? acc1[r] : -1e30f;
      float tmax = fmaxf(v0, v1);
#pragma unroll
      for (int m = 1; m < 16; m <<= 1) tmax = fmaxf(tmax, __shfl_xor(tmax, m, 32));
      float e = __expf(v0 - tmax) + __expf(v1 - tmax);
#pragma unroll
      for (int m = 1; m < 16; m <<= 1) e += __shfl_xor(e, m, 32);
      float nm = fmaxf(rmax[r], tmax);
      rsum[r] = rsum[r] * __expf(rmax[r] - nm) + e * __expf(tmax - nm);
      rmax[r] = nm;
    }
  }
  if (ln == 0) {
#pragma unroll
    for (int r = 0; r < 8; ++r) {
      red[wave][g * 8 + r][0] = rmax[r];
      red[wave][g * 8 + r][1] = rsum[r];
    }
  }
  __syncthreads();
  if (tid < 16) {                                   // merge 8 waves per row
    float m = -1e30f, s = 0.0f;
    for (int w = 0; w < 8; ++w) {
      float wm = red[w][tid][0], ws2 = red[w][tid][1];
      float nm = fmaxf(m, wm);
      s = s * __expf(m - nm) + ws2 * __expf(wm - nm);
      m = nm;
    }
    lse[tid] = m + __logf(s);
  }
  {                                                 // target score partial dots
    int row = tid >> 4, seg = tid & 15;
    int rglob = strip * 16 + row;
    int n = rglob / CT, t = rglob % CT;
    int y = captions[n * (CT + 1) + t + 1];
    const __bf16* wr = Wout_t + (size_t)y * CH;
    const __bf16* ar = As + (size_t)row * CH;
    float p = 0.0f;
    for (int k = seg * 32; k < seg * 32 + 32; ++k)
      p += (float)ar[k] * (float)wr[k];
    part[row][seg] = p;
  }
  __syncthreads();
  if (tid < 16) {
    int rglob = strip * 16 + tid;
    int n = rglob / CT, t = rglob % CT;
    int y = captions[n * (CT + 1) + t + 1];
    float sc = b_out[y];
    for (int s2 = 0; s2 < 16; ++s2) sc += part[tid][s2];
    float nll = lse[tid] - sc;
    rowNll[tid] = (y != 0) ? nll : 0.0f;            // NULL_IDX mask
  }
  __syncthreads();
  if (tid == 0) {
    float sum = 0.0f;
    for (int r = 0; r < 16; ++r) sum += rowNll[r];
    atomicAdd(loss, sum * (1.0f / (float)CN));
  }
}

// ---------------------------------------------------------------------------

extern "C" void kernel_launch(void* const* d_in, const int* in_sizes, int n_in,
                              void* d_out, int out_size, void* d_ws, size_t ws_size,
                              hipStream_t stream) {
  const float* features = (const float*)d_in[0];   // (N, D)
  const int*   captions = (const int*)  d_in[1];   // (N, T+1)
  const float* W_proj   = (const float*)d_in[2];   // (D, H)
  const float* b_proj   = (const float*)d_in[3];   // (H,)
  const float* W_embed  = (const float*)d_in[4];   // (V, W)
  const float* Wx       = (const float*)d_in[5];   // (W, H)
  const float* Wh       = (const float*)d_in[6];   // (H, H)
  const float* b        = (const float*)d_in[7];   // (H,)
  const float* W_out    = (const float*)d_in[8];   // (H, V)
  const float* b_out    = (const float*)d_in[9];   // (V,)
  float* loss = (float*)d_out;

  // Workspace layout (all offsets 256B aligned)
  char* ws = (char*)d_ws;
  __bf16* Wproj_t = (__bf16*)(ws + 0);                         // 512x1280 bf16
  __bf16* Wx_t    = (__bf16*)(ws + 1310720);                   // 512x256
  __bf16* Wh_t    = (__bf16*)(ws + 1572864);                   // 512x512
  __bf16* Wout_t  = (__bf16*)(ws + 2097152);                   // 10000x512
  float*  xWxb    = (float*) (ws + 12337152);                  // 16384x512 f32
  float*  hA      = (float*) (ws + 45891584);                  // 256x512 f32
  float*  hB      = (float*) (ws + 46415872);                  // 256x512 f32
  __bf16* hs_bf   = (__bf16*)(ws + 46940160);                  // 16384x512 bf16
  // total ~63.7 MB

  zero_f32_kernel<<<(out_size + 255) / 256, 256, 0, stream>>>(loss, out_size);

  // Weight transpose+convert (one-time):  out[N x K] = in[K x N]^T as bf16
  {
    long long e;
    e = (long long)CD * CH;   // W_proj -> Wproj_t
    transpose_to_bf16_kernel<<<(int)((e + 255) / 256), 256, 0, stream>>>(W_proj, Wproj_t, CD, CH);
    e = (long long)CW * CH;   // Wx -> Wx_t
    transpose_to_bf16_kernel<<<(int)((e + 255) / 256), 256, 0, stream>>>(Wx, Wx_t, CW, CH);
    e = (long long)CH * CH;   // Wh -> Wh_t
    transpose_to_bf16_kernel<<<(int)((e + 255) / 256), 256, 0, stream>>>(Wh, Wh_t, CH, CH);
    e = (long long)CH * CV;   // W_out -> Wout_t
    transpose_to_bf16_kernel<<<(int)((e + 255) / 256), 256, 0, stream>>>(W_out, Wout_t, CH, CV);
  }

  // h0 = features @ W_proj + b_proj    (256 x 1280 x 512) -> hA
  {
    int tiles = (CN / 16) * (CH / 16);              // 512 waves
    gemm_af32_btbf16_kernel<<<(tiles + 3) / 4, 128, 0, stream>>>(
        features, nullptr, Wproj_t, b_proj, hA, CN, CH, CD);
  }

  // xWxb = W_embed[captions_in] @ Wx + b   (16384 x 256 x 512)
  {
    int tiles = ((CN * CT) / 16) * (CH / 16);       // 32768 waves
    gemm_af32_btbf16_kernel<<<(tiles + 3) / 4, 128, 0, stream>>>(
        W_embed, captions, Wx_t, b, xWxb, CN * CT, CH, CW);
  }

  // RNN scan: 64 sequential steps (stream ordering provides the dependency)
  {
    float* hp = hA;
    float* hn = hB;
    for (int t = 0; t < CT; ++t) {
      rnn_step_kernel<<<128, 128, 0, stream>>>(hp, Wh_t, xWxb, hn, hs_bf, t);
      float* tmp = hp; hp = hn; hn = tmp;
    }
  }

  // Fused vocabulary GEMM + logsumexp + masked NLL  (dominant: ~168 GFLOP)
  scores_loss_kernel<<<(CN * CT) / 16, 256, 0, stream>>>(
      hs_bf, Wout_t, b_out, captions, loss);
}
